// TT_39118562132562
// MI455X (gfx1250) — compile-verified, matrix operand order checked by reading, not run on hardware
//
#include <hip/hip_runtime.h>
#include <math.h>

// B=64 batches, D=128 steps, A=128 states.
// out = [ log_px (64*128 floats, value replicated across the 128 cols) , log_p_a1 (128) ]

typedef __attribute__((ext_vector_type(16))) __bf16        v16bf;
typedef __attribute__((ext_vector_type(8)))  float         v8f;
typedef __attribute__((ext_vector_type(4)))  unsigned int  v4u;

union Frag {
    v16bf        v;
    v4u          q[2];
    unsigned int u[8];
};

__device__ __forceinline__ unsigned int pack_bf16(float a, float b) {
    unsigned int ua = __float_as_uint(a), ub = __float_as_uint(b);
    ua = (ua + 0x7FFFu + ((ua >> 16) & 1u)) >> 16;   // round-to-nearest-even
    ub = (ub + 0x7FFFu + ((ub >> 16) & 1u)) >> 16;
    return ua | (ub << 16);
}

// ---------------- preprocess 1: row logsumexp of transition logits ----------
__global__ __launch_bounds__(256) void k_lse_rows(const float* __restrict__ uT,
                                                  float* __restrict__ lseT,
                                                  int nrows) {
    int r = blockIdx.x * blockDim.x + threadIdx.x;
    if (r >= nrows) return;
    const float4* p4 = reinterpret_cast<const float4*>(uT + (size_t)r * 128);
    float m = -INFINITY;
    for (int i = 0; i < 32; ++i) {
        float4 v = p4[i];
        m = fmaxf(fmaxf(fmaxf(m, v.x), fmaxf(v.y, v.z)), v.w);
    }
    float s = 0.f;
    for (int i = 0; i < 32; ++i) {
        float4 v = p4[i];
        s += __expf(v.x - m) + __expf(v.y - m) + __expf(v.z - m) + __expf(v.w - m);
    }
    lseT[r] = m + __logf(s);
}

// ---------------- preprocess 2: emission tables + log_p_a1 ------------------
__global__ __launch_bounds__(256) void k_emis(const float* __restrict__ l,
                                              float* __restrict__ c0,
                                              float* __restrict__ c1,
                                              const float* __restrict__ u1,
                                              float* __restrict__ logp1,
                                              float* __restrict__ out_tail) {
    int i = blockIdx.x * blockDim.x + threadIdx.x;   // 0..16383
    float lv = l[i];
    float sp = (lv > 0.f) ? (lv + log1pf(__expf(-lv))) : log1pf(__expf(lv));
    c0[i] = -sp;
    c1[i] = lv - sp;
    if (blockIdx.x == 0) {
        __shared__ float lse1;
        if (threadIdx.x == 0) {
            float m = -INFINITY;
            for (int a = 0; a < 128; ++a) m = fmaxf(m, u1[a]);
            float s = 0.f;
            for (int a = 0; a < 128; ++a) s += __expf(u1[a] - m);
            lse1 = m + __logf(s);
        }
        __syncthreads();
        if (threadIdx.x < 128) {
            float v = u1[threadIdx.x] - lse1;
            logp1[threadIdx.x]    = v;
            out_tail[threadIdx.x] = v;
        }
    }
}

// ---------------- main sequential scan: one WGP, 8 wave32s ------------------
__global__ __launch_bounds__(256) void k_scan(const float* __restrict__ x,
                                              const float* __restrict__ uT,
                                              const float* __restrict__ lseT,
                                              const float* __restrict__ c0,
                                              const float* __restrict__ c1,
                                              const float* __restrict__ logp1,
                                              float* __restrict__ out) {
    // Fragment-line staging: for each (tile, lane) the lane's 8 frag dwords are
    // contiguous (32B, 32B-aligned) -> 2x ds_load_b128 per fragment, no shuffles.
    __shared__ float        V[64 * 128];             // fp32 state, 32 KB
    __shared__ unsigned int Af[2 * 4 * 32 * 8];      // A frag lines, 8 KB  (per K-half)
    __shared__ unsigned int Bf[2 * 8 * 32 * 8];      // B frag lines, 16 KB (per K-half)
    __shared__ float        pmax[256];
    __shared__ float        mrow[64];
    __shared__ float        xcol[64];

    const int tid  = threadIdx.x;
    const int wid  = tid >> 5;        // wave id = N-tile (0..7)
    const int lane = tid & 31;
    const int half = lane >> 4;
    const int l16  = lane & 15;
    const int jcol = wid * 16 + l16;  // this lane's output column

    // init: V[b,j] = px(b,0,j) + log_p_a1[j]
    for (int i = 0; i < 32; ++i) {
        int idx = i * 256 + tid;                 // 8192 elements
        int b = idx >> 7, j = idx & 127;
        float xv = x[b * 128];                   // x[b,0]
        V[idx] = (xv != 0.f ? c1[j] : c0[j]) + logp1[j];
    }
    __syncthreads();

    for (int t = 0; t < 127; ++t) {
        // prefetch next step's 64KB transition slab (global_prefetch_b8)
        if (t + 1 < 127) {
            const char* pf = (const char*)(uT + (size_t)(t + 1) * 16384);
            __builtin_prefetch(pf + tid * 256, 0, 1);
        }
        // per-row max, 2-level: 4 threads per row scan 32 elems (float4), then combine
        {
            int b = tid >> 2, part = tid & 3;
            const float4* vr = reinterpret_cast<const float4*>(&V[b * 128 + part * 32]);
            float4 v0 = vr[0];
            float m = fmaxf(fmaxf(v0.x, v0.y), fmaxf(v0.z, v0.w));
            for (int i = 1; i < 8; ++i) {
                float4 v = vr[i];
                m = fmaxf(fmaxf(fmaxf(m, v.x), fmaxf(v.y, v.z)), v.w);
            }
            pmax[tid] = m;
        }
        __syncthreads();
        if (tid < 64) {
            mrow[tid] = fmaxf(fmaxf(pmax[tid * 4], pmax[tid * 4 + 1]),
                              fmaxf(pmax[tid * 4 + 2], pmax[tid * 4 + 3]));
            xcol[tid] = x[tid * 128 + (t + 1)];
        }
        __syncthreads();

        v8f acc[4] = {};                         // 4 M-tiles for this wave's N-tile

        for (int kh = 0; kh < 2; ++kh) {         // K split in halves of 64
            // ---- write A fragment lines: one (kt,mt,lane) line per thread ----
            {
                int kt = tid >> 7;
                int mt = (tid >> 5) & 3;
                int ln = tid & 31;
                int hf = ln >> 4, s16 = ln & 15;
                int row   = mt * 16 + s16;
                int kbase = kh * 64 + kt * 32 + 8 * hf;
                float m = mrow[row];
                const float4* vq = reinterpret_cast<const float4*>(&V[row * 128 + kbase]);
                float4 a0 = vq[0], a1 = vq[1];   // k offsets 0..7   (r = 0..3)
                float4 b0 = vq[4], b1 = vq[5];   // k offsets 16..23 (r = 4..7)
                v4u lo, hi;
                lo[0] = pack_bf16(__expf(a0.x - m), __expf(a0.y - m));
                lo[1] = pack_bf16(__expf(a0.z - m), __expf(a0.w - m));
                lo[2] = pack_bf16(__expf(a1.x - m), __expf(a1.y - m));
                lo[3] = pack_bf16(__expf(a1.z - m), __expf(a1.w - m));
                hi[0] = pack_bf16(__expf(b0.x - m), __expf(b0.y - m));
                hi[1] = pack_bf16(__expf(b0.z - m), __expf(b0.w - m));
                hi[2] = pack_bf16(__expf(b1.x - m), __expf(b1.y - m));
                hi[3] = pack_bf16(__expf(b1.z - m), __expf(b1.w - m));
                v4u* dst = reinterpret_cast<v4u*>(&Af[((kt * 4 + mt) * 32 + ln) * 8]);
                dst[0] = lo;
                dst[1] = hi;
            }
            // ---- write B fragment lines: two (kt,nt,lane) lines per thread ----
            {
                const float* Tt = uT + (size_t)t * 16384;
                const float* Lt = lseT + t * 128;
                for (int li = 0; li < 2; ++li) {
                    int L  = li * 256 + tid;     // 0..511
                    int kt = L >> 8;
                    int nt = (L >> 5) & 7;
                    int ln = L & 31;
                    int hf = ln >> 4, s16 = ln & 15;
                    int col   = nt * 16 + s16;
                    int kbase = kh * 64 + kt * 32 + 8 * hf;
                    v4u lo, hi;
#pragma unroll
                    for (int g = 0; g < 2; ++g) {
#pragma unroll
                        for (int rr = 0; rr < 4; ++rr) {
                            int k = kbase + g * 16 + 2 * rr;
                            float e0 = __expf(Tt[k * 128 + col]       - Lt[k]);
                            float e1 = __expf(Tt[(k + 1) * 128 + col] - Lt[k + 1]);
                            if (g == 0) lo[rr] = pack_bf16(e0, e1);
                            else        hi[rr] = pack_bf16(e0, e1);
                        }
                    }
                    v4u* dst = reinterpret_cast<v4u*>(&Bf[((kt * 8 + nt) * 32 + ln) * 8]);
                    dst[0] = lo;
                    dst[1] = hi;
                }
            }
            __syncthreads();

            for (int kt = 0; kt < 2; ++kt) {     // two 16x16x32 K-chunks per half
                Frag bfm;
                const v4u* bp = reinterpret_cast<const v4u*>(
                    &Bf[((kt * 8 + wid) * 32 + lane) * 8]);
                bfm.q[0] = bp[0];
                bfm.q[1] = bp[1];
#pragma unroll
                for (int mt = 0; mt < 4; ++mt) {
                    Frag af;
                    const v4u* ap = reinterpret_cast<const v4u*>(
                        &Af[((kt * 4 + mt) * 32 + lane) * 8]);
                    af.q[0] = ap[0];
                    af.q[1] = ap[1];
                    acc[mt] = __builtin_amdgcn_wmma_f32_16x16x32_bf16(
                        false, af.v, false, bfm.v, (short)0, acc[mt], false, false);
                }
            }
            __syncthreads();
        }

        // V_new[b,j] = log(S) + m[b] + emission(b, t+1, j)
        float c0v = c0[(t + 1) * 128 + jcol];
        float c1v = c1[(t + 1) * 128 + jcol];
#pragma unroll
        for (int mt = 0; mt < 4; ++mt) {
#pragma unroll
            for (int r = 0; r < 8; ++r) {
                int b = mt * 16 + half * 8 + r;  // C layout: M = r + 8*half
                float s = acc[mt][r];
                float emis = (xcol[b] != 0.f) ? c1v : c0v;
                V[b * 128 + jcol] = __logf(fmaxf(s, 1e-30f)) + mrow[b] + emis;
            }
        }
        __syncthreads();
    }

    // final: L[b] = logsumexp_j V[b,j]; replicate across 128 output columns
    if (tid < 64) {
        const float* vr = &V[tid * 128];
        float m = vr[0];
        for (int j = 1; j < 128; ++j) m = fmaxf(m, vr[j]);
        float s = 0.f;
        for (int j = 0; j < 128; ++j) s += __expf(vr[j] - m);
        mrow[tid] = m + __logf(s);
    }
    __syncthreads();
    for (int i = 0; i < 32; ++i) {
        int idx = i * 256 + tid;                 // 8192 outputs
        out[idx] = mrow[idx >> 7];
    }
}

extern "C" void kernel_launch(void* const* d_in, const int* in_sizes, int n_in,
                              void* d_out, int out_size, void* d_ws, size_t ws_size,
                              hipStream_t stream) {
    const float* x  = (const float*)d_in[0];   // [64,128] binary
    const float* u1 = (const float*)d_in[1];   // [128]
    const float* uT = (const float*)d_in[2];   // [127,128,128]
    const float* l  = (const float*)d_in[3];   // [128,128]
    float* out = (float*)d_out;

    float* ws    = (float*)d_ws;
    float* lseT  = ws;                // 127*128 (pad to 16384)
    float* c0    = ws + 16384;        // 16384
    float* c1    = ws + 32768;        // 16384
    float* logp1 = ws + 49152;        // 128

    k_lse_rows<<<64, 256, 0, stream>>>(uT, lseT, 127 * 128);
    k_emis<<<64, 256, 0, stream>>>(l, c0, c1, u1, logp1, out + 64 * 128);
    k_scan<<<1, 256, 0, stream>>>(x, uT, lseT, c0, c1, logp1, out);
}
